// EncoderBlock_84997402788169
// MI455X (gfx1250) — matrix operand order repaired and checked by reading, not compiled, not run on hardware
//
#include <hip/hip_runtime.h>
#include <hip/hip_bf16.h>

typedef _Float16 f16;
typedef _Float16 v16h __attribute__((ext_vector_type(16)));
typedef _Float16 v8h  __attribute__((ext_vector_type(8)));
typedef float    v8f  __attribute__((ext_vector_type(8)));

#define DIM   512
#define WIN   8
#define HEADS 8
#define HD    64
#define EPS   1e-5f

#define QKV_E  (3 * DIM * DIM)   // 786432
#define PROJ_E (DIM * DIM)       // 262144
#define TOT_W  (2 * (QKV_E + PROJ_E))

#define ROWS_PER_BLK 64          // 4 window pairs = 4 WMMA M-tiles
#define PAIRS_PER_BLK 4
#define ASTR  520                // sA row stride (f16): 260 dwords = 4 banks skew/row
#define QSTR  1544               // sQKV row stride (f16): 772 dwords = 4 banks skew/row

// ---------------- weight f32 -> f16 conversion ----------------
__global__ __launch_bounds__(256) void cvt_weights(
    const float* __restrict__ q1, const float* __restrict__ p1,
    const float* __restrict__ q2, const float* __restrict__ p2,
    f16* __restrict__ o) {
  int i = blockIdx.x * blockDim.x + threadIdx.x;
  int stride = gridDim.x * blockDim.x;
  for (; i < TOT_W; i += stride) {
    float v;
    if (i < QKV_E)                       v = q1[i];
    else if (i < QKV_E + PROJ_E)         v = p1[i - QKV_E];
    else if (i < 2 * QKV_E + PROJ_E)     v = q2[i - (QKV_E + PROJ_E)];
    else                                 v = p2[i - (2 * QKV_E + PROJ_E)];
    o[i] = (f16)v;
  }
}

// Load a wave's A fragments (16 x 512 f16 tile, row-major stride ASTR in LDS)
// into registers once: 16 K-steps x v16h = 128 VGPRs.
// 16-bit A layout: lanes 0-15 hold M=lane, K {0..7, 16..23}+k0;
// lanes 16-31 hold the same rows with K offset +8.
__device__ __forceinline__ void load_afrags(const f16* __restrict__ sAbase,
                                            int lane, v16h* Af) {
  const int arow = lane & 15;
  const int akh  = (lane >> 4) * 8;
  const f16* ap = sAbase + arow * ASTR + akh;
#pragma unroll
  for (int kk = 0; kk < 16; ++kk) {
    v8h a0 = *(const v8h*)(ap + kk * 32);        // K = akh+0..7
    v8h a1 = *(const v8h*)(ap + kk * 32 + 16);   // K = akh+16..23
    v16h A;
#pragma unroll
    for (int i = 0; i < 8; ++i) { A[i] = a0[i]; A[8 + i] = a1[i]; }
    Af[kk] = A;
  }
}

// One 16x16 output tile vs register-resident A.
// B = W^T, W is (Nout x 512) row-major f16 in global (L2/WGP$-resident).
// 16-bit B layout: lanes 0-15 hold K {0..7, 16..23}, lanes 16-31 hold K +8,
// N = lane&15 -> per lane two 8-f16 chunks of one W row = 2 b128 loads.
__device__ __forceinline__ v8f gemm_regA(const v16h* Af,
                                         const f16* __restrict__ W,
                                         int ntile, int lane) {
  const int bn  = ntile * 16 + (lane & 15);
  const int bkh = (lane >> 4) * 8;
  const f16* bp = W + (long)bn * DIM + bkh;
  v8f acc{};
#pragma unroll
  for (int kk = 0; kk < 16; ++kk) {
    v8h b0 = *(const v8h*)(bp + kk * 32);
    v8h b1 = *(const v8h*)(bp + kk * 32 + 16);
    v16h B;
#pragma unroll
    for (int i = 0; i < 8; ++i) { B[i] = b0[i]; B[8 + i] = b1[i]; }
    acc = __builtin_amdgcn_wmma_f32_16x16x32_f16(
        false, Af[kk], false, B, (short)0, acc, false, false);
  }
  return acc;
}

// ---------------- fused window-attention layer ----------------
// One block (8 wave32s) = 4 window pairs = 64 rows. Waves with the same
// N-tile phase (0-3 even / 4-7 odd) stream identical B fragments -> WGP$
// reuse cuts L2 weight traffic 4x vs one-pair blocks.
// sQKV/sA are row-major with +4-dword skewed strides so 16 consecutive
// rows land on distinct LDS bank groups (no 16-way conflicts).
__global__ __launch_bounds__(256) void swin_layer(
    const float* __restrict__ xin, float* __restrict__ xout,
    const f16* __restrict__ wqkv, const float* __restrict__ bqkv,
    const f16* __restrict__ wproj, const float* __restrict__ bproj,
    const float* __restrict__ lnw, const float* __restrict__ lnb) {
  __shared__ f16 sQKV[ROWS_PER_BLK * QSTR];   // ~193KB row-major QKV (skewed)
  __shared__ f16 sA[ROWS_PER_BLK * ASTR];     // ~65KB A (LN out / attn out)

  const int tid  = threadIdx.x;
  const int lane = tid & 31;
  const int wid  = tid >> 5;
  const long rowbase = (long)blockIdx.x * ROWS_PER_BLK;

  // ---- Phase 1: LayerNorm over (8,512), one wave per window ----
  {
    const int w = wid;                 // window 0..7 within block
    const int l = lane;
    const float* src = xin + (rowbase + (long)w * 8) * DIM;  // window contiguous
    float s = 0.f, ss = 0.f;
#pragma unroll
    for (int j = 0; j < 16; ++j) {     // 8 elems/thread/step, coalesced
      int e = j * 256 + l * 8;
      float4 a = *(const float4*)(src + e);
      float4 b = *(const float4*)(src + e + 4);
      s  += a.x + a.y + a.z + a.w + b.x + b.y + b.z + b.w;
      ss += a.x*a.x + a.y*a.y + a.z*a.z + a.w*a.w
          + b.x*b.x + b.y*b.y + b.z*b.z + b.w*b.w;
    }
#pragma unroll
    for (int m = 16; m >= 1; m >>= 1) {          // wave32 reduction
      s  += __shfl_xor(s,  m, 32);
      ss += __shfl_xor(ss, m, 32);
    }
    const float mean = s * (1.f / 4096.f);
    const float var  = ss * (1.f / 4096.f) - mean * mean;
    const float rstd = rsqrtf(var + EPS);
#pragma unroll 4
    for (int j = 0; j < 16; ++j) {     // second pass: hits WGP$/L2
      int e = j * 256 + l * 8;
      float4 a = *(const float4*)(src + e);
      float4 b = *(const float4*)(src + e + 4);
      int r = e >> 9, c = e & 511;     // chunk never crosses a row
      const float* gw = lnw + r * DIM + c;
      const float* gb = lnb + r * DIM + c;
      v8h pk;
      pk[0] = (f16)((a.x - mean) * rstd * gw[0] + gb[0]);
      pk[1] = (f16)((a.y - mean) * rstd * gw[1] + gb[1]);
      pk[2] = (f16)((a.z - mean) * rstd * gw[2] + gb[2]);
      pk[3] = (f16)((a.w - mean) * rstd * gw[3] + gb[3]);
      pk[4] = (f16)((b.x - mean) * rstd * gw[4] + gb[4]);
      pk[5] = (f16)((b.y - mean) * rstd * gw[5] + gb[5]);
      pk[6] = (f16)((b.z - mean) * rstd * gw[6] + gb[6]);
      pk[7] = (f16)((b.w - mean) * rstd * gw[7] + gb[7]);
      *(v8h*)(sA + (w * 8 + r) * ASTR + c) = pk;
    }
  }
  __syncthreads();

  // ---- Phase 2: QKV GEMM (64x512) x (512x1536) -> f16 row-major LDS ----
  {
    const int mt = wid & 3;            // M-tile (window pair) 0..3
    const int g2 = wid >> 2;           // N-phase 0/1 (shared B stream per phase)
    v16h Af[16];
    load_afrags(sA + mt * 16 * ASTR, lane, Af);
    const int n  = (lane & 15);
    const int mh = (lane >> 4) * 8;    // C layout: VGPR r -> row r / r+8
    for (int nt = g2; nt < 96; nt += 2) {
      v8f acc = gemm_regA(Af, wqkv, nt, lane);
      const int ng = nt * 16 + n;
      const float bias = bqkv[ng];
#pragma unroll
      for (int r = 0; r < 8; ++r)
        sQKV[(mt * 16 + r + mh) * QSTR + ng] = (f16)(acc[r] + bias);
    }
  }
  __syncthreads();

  // ---- Phase 3: per-head 8x8 attention (vectorized b128 LDS reads) ----
#pragma unroll 1
  for (int half = 0; half < 2; ++half) {
    const int job = half * 32 + (tid >> 3);   // 64 jobs = 4 pairs x 2 win x 8 heads
    const int i   = tid & 7;                  // query row
    const int p   = job >> 4;
    const int w   = (job >> 3) & 1;
    const int h   = job & 7;
    const int tr  = p * 16 + w * 8;           // window row base within block
    const f16* qp = sQKV + (tr + i) * QSTR + h * HD;
    float qf[HD];
#pragma unroll
    for (int j8 = 0; j8 < 8; ++j8) {          // q: 8 x ds_load_b128, cvt once
      v8h qv = *(const v8h*)(qp + j8 * 8);
#pragma unroll
      for (int q = 0; q < 8; ++q) qf[j8 * 8 + q] = (float)qv[q];
    }
    float sc[8];
    float mx = -1e30f;
#pragma unroll
    for (int m = 0; m < 8; ++m) {
      const f16* kp = sQKV + (tr + m) * QSTR + DIM + h * HD;
      float d = 0.f;
#pragma unroll
      for (int j8 = 0; j8 < 8; ++j8) {
        v8h kv = *(const v8h*)(kp + j8 * 8);
#pragma unroll
        for (int q = 0; q < 8; ++q) d = fmaf(qf[j8 * 8 + q], (float)kv[q], d);
      }
      sc[m] = d * 0.125f;                     // hd^-0.5 = 1/8
      mx = fmaxf(mx, sc[m]);
    }
    float ssum = 0.f;
#pragma unroll
    for (int m = 0; m < 8; ++m) { sc[m] = __expf(sc[m] - mx); ssum += sc[m]; }
    const float inv = 1.f / ssum;
    f16* op = sA + (tr + i) * ASTR + h * HD;
#pragma unroll
    for (int d8 = 0; d8 < 8; ++d8) {          // out: 64 b128 loads + 8 b128 stores
      float ov[8] = {0.f, 0.f, 0.f, 0.f, 0.f, 0.f, 0.f, 0.f};
#pragma unroll
      for (int m = 0; m < 8; ++m) {
        v8h vv = *(const v8h*)(sQKV + (tr + m) * QSTR + 2 * DIM + h * HD + d8 * 8);
#pragma unroll
        for (int q = 0; q < 8; ++q) ov[q] = fmaf(sc[m], (float)vv[q], ov[q]);
      }
      v8h pk;
#pragma unroll
      for (int q = 0; q < 8; ++q) pk[q] = (f16)(ov[q] * inv);
      *(v8h*)(op + d8 * 8) = pk;
    }
  }
  __syncthreads();

  // ---- Phase 4: projection GEMM (64x512) x (512x512) -> global f32 ----
  {
    const int mt = wid & 3;
    const int g2 = wid >> 2;
    v16h Af[16];
    load_afrags(sA + mt * 16 * ASTR, lane, Af);
    const int n  = (lane & 15);
    const int mh = (lane >> 4) * 8;
    for (int nt = g2; nt < 32; nt += 2) {
      v8f acc = gemm_regA(Af, wproj, nt, lane);
      const int ng = nt * 16 + n;
      const float bias = bproj[ng];
#pragma unroll
      for (int r = 0; r < 8; ++r)
        xout[(rowbase + mt * 16 + r + mh) * DIM + ng] = acc[r] + bias;
    }
  }
}

extern "C" void kernel_launch(void* const* d_in, const int* in_sizes, int n_in,
                              void* d_out, int out_size, void* d_ws, size_t ws_size,
                              hipStream_t stream) {
  (void)in_sizes; (void)n_in; (void)out_size; (void)ws_size;
  const float* x       = (const float*)d_in[0];
  const float* ln1_w   = (const float*)d_in[1];
  const float* ln1_b   = (const float*)d_in[2];
  const float* ln3_w   = (const float*)d_in[3];
  const float* ln3_b   = (const float*)d_in[4];
  const float* qkv1_w  = (const float*)d_in[5];
  const float* qkv1_b  = (const float*)d_in[6];
  const float* proj1_w = (const float*)d_in[7];
  const float* proj1_b = (const float*)d_in[8];
  const float* qkv2_w  = (const float*)d_in[9];
  const float* qkv2_b  = (const float*)d_in[10];
  const float* proj2_w = (const float*)d_in[11];
  const float* proj2_b = (const float*)d_in[12];
  float* out = (float*)d_out;

  f16* wbuf    = (f16*)d_ws;
  f16* w_qkv1  = wbuf;
  f16* w_proj1 = wbuf + QKV_E;
  f16* w_qkv2  = wbuf + QKV_E + PROJ_E;
  f16* w_proj2 = wbuf + 2 * QKV_E + PROJ_E;

  cvt_weights<<<512, 256, 0, stream>>>(qkv1_w, proj1_w, qkv2_w, proj2_w, wbuf);

  const int nblocks = (4 * 32768) / ROWS_PER_BLK;   // 2048 blocks of 64 rows
  swin_layer<<<nblocks, 256, 0, stream>>>(x,   out, w_qkv1, qkv1_b, w_proj1, proj1_b, ln1_w, ln1_b);
  swin_layer<<<nblocks, 256, 0, stream>>>(out, out, w_qkv2, qkv2_b, w_proj2, proj2_b, ln3_w, ln3_b);
}
